// RegionProposalNetwork_47330539602442
// MI455X (gfx1250) — matrix operand order repaired and checked by reading, not compiled, not run on hardware
//
#include <hip/hip_runtime.h>
#include <hip/hip_bf16.h>

// ---------------------------------------------------------------------------
// Types for CDNA5 WMMA
// ---------------------------------------------------------------------------
typedef __attribute__((ext_vector_type(16))) __bf16 v16bf;
typedef __attribute__((ext_vector_type(8)))  float  v8f;

union FragBF {
    v16bf v;
    uint4 q[2];
    unsigned short s[16];
};

#define HH 200
#define WW 200
#define CC 256
#define AA 9
#define NANCH (HH * WW * AA)   // 360000
#define PRE_NMS 12000
#define POST_NMS 2000
#define SORT_N1 524288         // next pow2 >= 360000
#define SORT_N2 16384          // next pow2 >= 12000
#define BBOX_CLIP 4.1351665567f
#define IMG 800.0f

__device__ __forceinline__ unsigned short f2bf(float f) {
    unsigned int u = __float_as_uint(f);
    unsigned int r = u + 0x7FFFu + ((u >> 16) & 1u);
    return (unsigned short)(r >> 16);
}

__device__ __forceinline__ unsigned int ordkey(float f) {
    unsigned int u = __float_as_uint(f);
    return (u & 0x80000000u) ? ~u : (u | 0x80000000u);
}

__device__ __forceinline__ void get_anchor(int idx, float& x0, float& y0,
                                           float& x1, float& y1) {
    int a = idx % AA;
    int p = idx / AA;
    int x = p % WW;
    int y = p / WW;
    int r = a / 3, s = a % 3;
    float sc = (s == 0) ? 32.f : ((s == 1) ? 64.f : 128.f);
    float sq = (r == 0) ? 0.70710678118654752f
                        : ((r == 1) ? 1.f : 1.41421356237309505f);
    float h = sc * sq;
    float w = sc / sq;
    float cx = ((float)x + 0.5f) * 4.f;
    float cy = ((float)y + 0.5f) * 4.f;
    x0 = cx - 0.5f * w; y0 = cy - 0.5f * h;
    x1 = cx + 0.5f * w; y1 = cy + 0.5f * h;
}

// ---------------------------------------------------------------------------
// Weight packing: conv_w (256,256,3,3) fp32 -> B-fragment bf16 layout
// wpk[t][kb][nt][lane][16]  (t=tap 0..8, kb=K-chunk 0..7, nt=oc-tile 0..15)
// B mapping (wave32): n = lane&15, k = elem + ((lane>>4)&1)*16
// ---------------------------------------------------------------------------
__global__ void pack_w_kernel(const float* __restrict__ conv_w,
                              unsigned short* __restrict__ wpk) {
    int o = blockIdx.x * blockDim.x + threadIdx.x;
    if (o >= 9 * 8 * 16 * 32 * 16) return;
    int e    = o & 15;
    int lane = (o >> 4) & 31;
    int nt   = (o >> 9) & 15;
    int kb   = (o >> 13) & 7;
    int t    = o >> 16;
    int k  = e + ((lane >> 4) & 1) * 16;
    int ic = kb * 32 + k;
    int oc = nt * 16 + (lane & 15);
    int ky = t / 3, kx = t % 3;
    float w = conv_w[((oc * CC + ic) * 3 + ky) * 3 + kx];
    wpk[o] = f2bf(w);
}

// cls_w (9,256) + reg_w (36,256) -> 48 padded outputs, 3 N-tiles
__global__ void pack_w2_kernel(const float* __restrict__ cls_w,
                               const float* __restrict__ reg_w,
                               unsigned short* __restrict__ w2pk) {
    int o = blockIdx.x * blockDim.x + threadIdx.x;
    if (o >= 8 * 3 * 32 * 16) return;
    int e    = o & 15;
    int lane = (o >> 4) & 31;
    int rest = o >> 9;
    int nt = rest % 3;
    int kb = rest / 3;
    int k  = e + ((lane >> 4) & 1) * 16;
    int ic = kb * 32 + k;
    int ch = nt * 16 + (lane & 15);
    float w = 0.f;
    if (ch < 9)       w = cls_w[ch * CC + ic];
    else if (ch < 45) w = reg_w[(ch - 9) * CC + ic];
    w2pk[o] = f2bf(w);
}

// ---------------------------------------------------------------------------
// Conv3x3 + ReLU + fused 1x1 heads via WMMA bf16 (f32 accumulate)
// grid = 2600 blocks (200 rows x 13 x-tiles of 16), 256 threads (8 waves)
// ---------------------------------------------------------------------------
__global__ __launch_bounds__(256) void conv_rpn_kernel(
    const float* __restrict__ feat,
    const unsigned short* __restrict__ wpk,
    const unsigned short* __restrict__ w2pk,
    const float* __restrict__ conv_b,
    const float* __restrict__ cls_b,
    const float* __restrict__ reg_b,
    float* __restrict__ cls_logits,
    float* __restrict__ regression) {
    __shared__ __align__(16) unsigned short s_in[3 * 18 * 256];
    __shared__ __align__(16) unsigned short s_t[16 * 256];

    int tile = blockIdx.x;
    int y = tile / 13, xt = tile % 13;
    int xbase = xt * 16;
    int tid = threadIdx.x;

    // Stage halo patch (3 rows x 18 cols x 256 ch), SAME zero padding
    for (int e = tid; e < 3 * 18 * 256; e += 256) {
        int c  = e / 54;
        int rc = e % 54;
        int r = rc / 18, col = rc % 18;
        int ry = y + r - 1;
        int cx = xbase + col - 1;
        float v = 0.f;
        if (ry >= 0 && ry < HH && cx >= 0 && cx < WW)
            v = feat[c * (HH * WW) + ry * WW + cx];
        s_in[rc * 256 + c] = f2bf(v);
    }
    __syncthreads();

    int lane = tid & 31;
    int wv   = tid >> 5;
    int M    = lane & 15;
    int sel  = (lane & 16) ? 16 : 0;
    int mrow = (lane & 16) ? 8 : 0;

    v8f acc0 = {};
    v8f acc1 = {};

    for (int t = 0; t < 9; ++t) {
        int ky = t / 3, kx = t % 3;
        int rcbase = (ky * 18 + M + kx) * 256;
        for (int kb = 0; kb < 8; ++kb) {
            int cb = kb * 32;
            FragBF a;
            a.q[0] = *(const uint4*)&s_in[rcbase + cb + sel];
            a.q[1] = *(const uint4*)&s_in[rcbase + cb + sel + 8];
            const unsigned short* wb = wpk + (size_t)((t * 8 + kb) * 16) * 512;
            FragBF b0, b1;
            const uint4* w0 = (const uint4*)(wb + (2 * wv) * 512 + lane * 16);
            b0.q[0] = w0[0]; b0.q[1] = w0[1];
            const uint4* w1 = (const uint4*)(wb + (2 * wv + 1) * 512 + lane * 16);
            b1.q[0] = w1[0]; b1.q[1] = w1[1];
            acc0 = __builtin_amdgcn_wmma_f32_16x16x32_bf16(
                false, a.v, false, b0.v, (short)0, acc0, false, false);
            acc1 = __builtin_amdgcn_wmma_f32_16x16x32_bf16(
                false, a.v, false, b1.v, (short)0, acc1, false, false);
        }
    }

    // bias + ReLU -> bf16 tile t in LDS (16 positions x 256 channels)
    for (int v = 0; v < 8; ++v) {
        int mpos = v + mrow;
        int oc0 = (2 * wv) * 16 + M;
        float t0 = fmaxf(acc0[v] + conv_b[oc0], 0.f);
        s_t[mpos * 256 + oc0] = f2bf(t0);
        int oc1 = (2 * wv + 1) * 16 + M;
        float t1 = fmaxf(acc1[v] + conv_b[oc1], 0.f);
        s_t[mpos * 256 + oc1] = f2bf(t1);
    }
    __syncthreads();

    // fused 1x1 heads: N = 48 (9 cls + 36 reg + 3 pad) -> 3 wave tiles
    if (wv < 3) {
        v8f acc2 = {};
        for (int kb = 0; kb < 8; ++kb) {
            int cb = kb * 32;
            FragBF a;
            a.q[0] = *(const uint4*)&s_t[M * 256 + cb + sel];
            a.q[1] = *(const uint4*)&s_t[M * 256 + cb + sel + 8];
            FragBF b;
            const uint4* wq = (const uint4*)(w2pk + (kb * 3 + wv) * 512 + lane * 16);
            b.q[0] = wq[0]; b.q[1] = wq[1];
            acc2 = __builtin_amdgcn_wmma_f32_16x16x32_bf16(
                false, a.v, false, b.v, (short)0, acc2, false, false);
        }
        for (int v = 0; v < 8; ++v) {
            int mpos = v + mrow;
            int x = xbase + mpos;
            if (x < WW) {
                int p = y * WW + x;
                int ch = wv * 16 + M;
                float val = acc2[v];
                if (ch < 9) {
                    val += cls_b[ch];
                    cls_logits[p * 9 + ch] = 1.f / (1.f + expf(-val));
                } else if (ch < 45) {
                    val += reg_b[ch - 9];
                    regression[p * 36 + (ch - 9)] = val;
                }
            }
        }
    }
}

// ---------------------------------------------------------------------------
// Generic bitonic sort stage (descending by uint key, int payload)
// ---------------------------------------------------------------------------
__global__ void bitonic_stage(unsigned int* key, int* val, int n, int k, int j) {
    int i = blockIdx.x * blockDim.x + threadIdx.x;
    if (i >= n) return;
    int ixj = i ^ j;
    if (ixj > i) {
        unsigned int a = key[i], b = key[ixj];
        bool up = ((i & k) == 0);
        if ((up && a < b) || (!up && a > b)) {
            key[i] = b; key[ixj] = a;
            int t = val[i]; val[i] = val[ixj]; val[ixj] = t;
        }
    }
}

__global__ void build_score_keys(const float* __restrict__ cls,
                                 unsigned int* key, int* val) {
    int i = blockIdx.x * blockDim.x + threadIdx.x;
    if (i >= SORT_N1) return;
    if (i < NANCH) { key[i] = ordkey(cls[i]); val[i] = i; }
    else           { key[i] = 0u;            val[i] = -1; }
}

// ---------------------------------------------------------------------------
// Decode top-12000, clip, min-size filter
// ---------------------------------------------------------------------------
__global__ void decode_kernel(const int* __restrict__ val1,
                              const float* __restrict__ cls,
                              const float* __restrict__ reg,
                              float4* __restrict__ boxes,
                              unsigned int* __restrict__ key2,
                              int* __restrict__ val2) {
    int i = blockIdx.x * blockDim.x + threadIdx.x;
    if (i >= SORT_N2) return;
    if (i >= PRE_NMS) { key2[i] = 0u; val2[i] = 0; return; }
    int a = val1[i];
    float score = cls[a];
    float ax0, ay0, ax1, ay1;
    get_anchor(a, ax0, ay0, ax1, ay1);
    float wa = ax1 - ax0, ha = ay1 - ay0;
    float cxa = ax0 + 0.5f * wa, cya = ay0 + 0.5f * ha;
    int p = a / AA, an = a % AA;
    int bidx = p * 36 + 4 * an;
    float dx = reg[bidx + 0];
    float dy = reg[bidx + 1];
    float dw = fminf(reg[bidx + 2], BBOX_CLIP);
    float dh = fminf(reg[bidx + 3], BBOX_CLIP);
    float cx = dx * wa + cxa;
    float cy = dy * ha + cya;
    float w = expf(dw) * wa;
    float h = expf(dh) * ha;
    float x0 = fminf(fmaxf(cx - 0.5f * w, 0.f), IMG);
    float y0 = fminf(fmaxf(cy - 0.5f * h, 0.f), IMG);
    float x1 = fminf(fmaxf(cx + 0.5f * w, 0.f), IMG);
    float y1 = fminf(fmaxf(cy + 0.5f * h, 0.f), IMG);
    bool valid = ((x1 - x0) >= 1.f) && ((y1 - y0) >= 1.f);
    boxes[i] = make_float4(x0, y0, x1, y1);
    key2[i] = valid ? ordkey(score) : 0u;
    val2[i] = i;
}

__global__ void gather_obox(const float4* __restrict__ boxes,
                            const int* __restrict__ val2,
                            float4* __restrict__ obox) {
    int r = blockIdx.x * blockDim.x + threadIdx.x;
    if (r >= SORT_N2) return;
    int v = val2[r];
    if (v < 0 || v >= PRE_NMS) v = 0;
    obox[r] = boxes[v];
}

// ---------------------------------------------------------------------------
// Greedy NMS (single block), boxes pre-sorted by score descending
// ---------------------------------------------------------------------------
__global__ __launch_bounds__(1024) void nms_kernel(
    const float4* __restrict__ obox,
    const unsigned int* __restrict__ okey,
    unsigned char* __restrict__ alive_g) {
    __shared__ unsigned char al[PRE_NMS];
    int tid = threadIdx.x, nt = blockDim.x;
    for (int i = tid; i < PRE_NMS; i += nt) al[i] = (okey[i] > 0u) ? 1 : 0;
    __syncthreads();
    for (int i = 0; i < PRE_NMS; ++i) {
        if (al[i]) {
            float4 bi = obox[i];
            float ai = (bi.z - bi.x) * (bi.w - bi.y);
            for (int j = i + 1 + tid; j < PRE_NMS; j += nt) {
                if (al[j]) {
                    float4 bj = obox[j];
                    float lx = fmaxf(bi.x, bj.x), ly = fmaxf(bi.y, bj.y);
                    float rx = fminf(bi.z, bj.z), ry = fminf(bi.w, bj.w);
                    float iw = fmaxf(rx - lx, 0.f), ih = fmaxf(ry - ly, 0.f);
                    float inter = iw * ih;
                    float aj = (bj.z - bj.x) * (bj.w - bj.y);
                    float iou = inter / (ai + aj - inter);
                    if (iou > 0.7f) al[j] = 0;
                }
            }
        }
        __syncthreads();
    }
    for (int i = tid; i < PRE_NMS; i += nt) alive_g[i] = al[i];
}

__global__ void gather_proposals(const float4* __restrict__ obox,
                                 const unsigned char* __restrict__ alive,
                                 const unsigned int* __restrict__ okey,
                                 float4* __restrict__ out) {
    __shared__ int list[POST_NMS];
    if (threadIdx.x == 0) {
        int c = 0;
        for (int r = 0; r < PRE_NMS && c < POST_NMS; ++r)
            if (alive[r] && okey[r] > 0u) list[c++] = r;
        for (int r = 0; r < PRE_NMS && c < POST_NMS; ++r)
            if (!(alive[r] && okey[r] > 0u)) list[c++] = r;
    }
    __syncthreads();
    for (int i = threadIdx.x; i < POST_NMS; i += blockDim.x)
        out[i] = obox[list[i]];
}

// ---------------------------------------------------------------------------
// Anchor/GT IoU + labeling
// ---------------------------------------------------------------------------
__device__ __forceinline__ float iou_ag(float ax0, float ay0, float ax1,
                                        float ay1, float aa, float4 G) {
    float ga = (G.z - G.x) * (G.w - G.y);
    float lx = fmaxf(G.x, ax0), ly = fmaxf(G.y, ay0);
    float rx = fminf(G.z, ax1), ry = fminf(G.w, ay1);
    float iw = fmaxf(rx - lx, 0.f), ih = fmaxf(ry - ly, 0.f);
    float inter = iw * ih;
    return inter / (ga + aa - inter);
}

__global__ void iou_pass1(const float4* __restrict__ gtb,
                          float* __restrict__ m_val,
                          int* __restrict__ m_idx,
                          unsigned int* __restrict__ gmax) {
    int a = blockIdx.x * blockDim.x + threadIdx.x;
    if (a >= NANCH) return;
    float x0, y0, x1, y1;
    get_anchor(a, x0, y0, x1, y1);
    float aa = (x1 - x0) * (y1 - y0);
    float best = -1.f;
    int bg = 0;
    for (int g = 0; g < 32; ++g) {
        float iou = iou_ag(x0, y0, x1, y1, aa, gtb[g]);
        if (iou > best) { best = iou; bg = g; }
        atomicMax(&gmax[g], __float_as_uint(iou));
    }
    m_val[a] = best;
    m_idx[a] = bg;
}

__global__ void label_pass(const float4* __restrict__ gtb,
                           const unsigned int* __restrict__ gmax,
                           const float* __restrict__ m_val,
                           float* __restrict__ label,
                           int* __restrict__ posCount) {
    int a = blockIdx.x * blockDim.x + threadIdx.x;
    if (a >= NANCH) return;
    float x0, y0, x1, y1;
    get_anchor(a, x0, y0, x1, y1);
    float aa = (x1 - x0) * (y1 - y0);
    bool lq = false;
    for (int g = 0; g < 32; ++g) {
        float iou = iou_ag(x0, y0, x1, y1, aa, gtb[g]);
        if (__float_as_uint(iou) == gmax[g]) lq = true;
    }
    float mv = m_val[a];
    float lab = (mv >= 0.7f) ? 1.f : ((mv < 0.3f) ? 0.f : -1.f);
    if (lq) lab = 1.f;
    label[a] = lab;
    if (lab == 1.f) atomicAdd(posCount, 1);
}

__global__ void build_sample_keys(const float* __restrict__ rnd,
                                  const float* __restrict__ label,
                                  unsigned int* kp, int* vp,
                                  unsigned int* kn, int* vn) {
    int i = blockIdx.x * blockDim.x + threadIdx.x;
    if (i >= SORT_N1) return;
    if (i < NANCH) {
        float lab = label[i];
        float r = rnd[i];
        kp[i] = ordkey((lab == 1.f) ? r : -1.f);
        kn[i] = ordkey((lab == 0.f) ? r : -1.f);
        vp[i] = i; vn[i] = i;
    } else {
        kp[i] = 0u; kn[i] = 0u; vp[i] = -1; vn[i] = -1;
    }
}

// ---------------------------------------------------------------------------
// Losses
// ---------------------------------------------------------------------------
__global__ void loss_kernel(const float* __restrict__ cls,
                            const float* __restrict__ reg,
                            const float4* __restrict__ gtb,
                            const int* __restrict__ m_idx,
                            const int* __restrict__ vpos,
                            const int* __restrict__ vneg,
                            const int* __restrict__ posCount,
                            float* __restrict__ out_losses) {
    int tid = threadIdx.x;
    int P = *posCount;
    int np = min(P, 128);
    int nn = 256 - np;
    const float eps = 1e-7f;
    if (tid < 128) {
        if (tid < np) {
            int a = vpos[tid];
            if (a >= 0) {
                float p = fminf(fmaxf(cls[a], eps), 1.f - eps);
                atomicAdd(&out_losses[0], -logf(p) * (1.f / 256.f));
                // regression target
                float ax0, ay0, ax1, ay1;
                get_anchor(a, ax0, ay0, ax1, ay1);
                float wa = ax1 - ax0, ha = ay1 - ay0;
                float cxa = ax0 + 0.5f * wa, cya = ay0 + 0.5f * ha;
                float4 G = gtb[m_idx[a]];
                float wg = G.z - G.x, hg = G.w - G.y;
                float cxg = G.x + 0.5f * wg, cyg = G.y + 0.5f * hg;
                float t0 = (cxg - cxa) / wa;
                float t1 = (cyg - cya) / ha;
                float t2 = logf(wg / wa);
                float t3 = logf(hg / ha);
                int p2 = a / AA, an = a % AA;
                int b = p2 * 36 + 4 * an;
                float rl = fabsf(reg[b] - t0) + fabsf(reg[b + 1] - t1) +
                           fabsf(reg[b + 2] - t2) + fabsf(reg[b + 3] - t3);
                atomicAdd(&out_losses[1], rl * (1.f / 256.f));
            }
        }
    } else if (tid < 384) {
        int i = tid - 128;
        if (i < nn) {
            int a = vneg[i];
            if (a >= 0) {
                float p = fminf(fmaxf(cls[a], eps), 1.f - eps);
                atomicAdd(&out_losses[0], -logf(1.f - p) * (1.f / 256.f));
            }
        }
    }
}

__global__ void init_kernel(unsigned int* gmax, int* posCount,
                            float* out_losses) {
    int i = threadIdx.x;
    if (i < 32) gmax[i] = 0u;
    if (i == 32) *posCount = 0;
    if (i == 33) { out_losses[0] = 0.f; out_losses[1] = 0.f; }
}

// ---------------------------------------------------------------------------
// Host orchestration
// ---------------------------------------------------------------------------
static void bitonic_sort(unsigned int* key, int* val, int n,
                         hipStream_t stream) {
    int threads = 256, blocks = (n + threads - 1) / threads;
    for (int k = 2; k <= n; k <<= 1)
        for (int j = k >> 1; j > 0; j >>= 1)
            bitonic_stage<<<blocks, threads, 0, stream>>>(key, val, n, k, j);
}

extern "C" void kernel_launch(void* const* d_in, const int* in_sizes, int n_in,
                              void* d_out, int out_size, void* d_ws,
                              size_t ws_size, hipStream_t stream) {
    const float* features = (const float*)d_in[0];
    const float* gt_box   = (const float*)d_in[1];
    const float* smp_rand = (const float*)d_in[2];
    const float* conv_w   = (const float*)d_in[3];
    const float* conv_b   = (const float*)d_in[4];
    const float* cls_w    = (const float*)d_in[5];
    const float* cls_b    = (const float*)d_in[6];
    const float* reg_w    = (const float*)d_in[7];
    const float* reg_b    = (const float*)d_in[8];
    float* out = (float*)d_out;

    // workspace carving (256B aligned)
    char* base = (char*)d_ws;
    size_t off = 0;
    auto carve = [&](size_t bytes) {
        char* p = base + off;
        off = (off + bytes + 255) & ~(size_t)255;
        return p;
    };
    unsigned short* wpk  = (unsigned short*)carve(589824 * 2);
    unsigned short* w2pk = (unsigned short*)carve(12288 * 2);
    float* cls_logits = (float*)carve((size_t)NANCH * 4);
    float* regression = (float*)carve((size_t)NANCH * 4 * 4);
    unsigned int* key1 = (unsigned int*)carve((size_t)SORT_N1 * 4);
    int*          val1 = (int*)carve((size_t)SORT_N1 * 4);
    float4* boxes = (float4*)carve((size_t)PRE_NMS * 16);
    unsigned int* key2 = (unsigned int*)carve((size_t)SORT_N2 * 4);
    int*          val2 = (int*)carve((size_t)SORT_N2 * 4);
    float4* obox = (float4*)carve((size_t)SORT_N2 * 16);
    unsigned char* alive = (unsigned char*)carve(SORT_N2);
    float* m_val = (float*)carve((size_t)NANCH * 4);
    int*   m_idx = (int*)carve((size_t)NANCH * 4);
    float* label = (float*)carve((size_t)NANCH * 4);
    unsigned int* gmax = (unsigned int*)carve(128);
    int* posCount = (int*)carve(64);
    unsigned int* keyP = (unsigned int*)carve((size_t)SORT_N1 * 4);
    int*          valP = (int*)carve((size_t)SORT_N1 * 4);
    unsigned int* keyN = (unsigned int*)carve((size_t)SORT_N1 * 4);
    int*          valN = (int*)carve((size_t)SORT_N1 * 4);
    (void)ws_size; (void)in_sizes; (void)n_in; (void)out_size;

    init_kernel<<<1, 64, 0, stream>>>(gmax, posCount, out + 8000);

    pack_w_kernel<<<(589824 + 255) / 256, 256, 0, stream>>>(conv_w, wpk);
    pack_w2_kernel<<<(12288 + 255) / 256, 256, 0, stream>>>(cls_w, reg_w, w2pk);

    conv_rpn_kernel<<<2600, 256, 0, stream>>>(features, wpk, w2pk, conv_b,
                                              cls_b, reg_b, cls_logits,
                                              regression);

    // top-12000 + NMS ordering via full descending sort of scores
    build_score_keys<<<SORT_N1 / 256, 256, 0, stream>>>(cls_logits, key1, val1);
    bitonic_sort(key1, val1, SORT_N1, stream);

    decode_kernel<<<SORT_N2 / 256, 256, 0, stream>>>(val1, cls_logits,
                                                     regression, boxes, key2,
                                                     val2);
    bitonic_sort(key2, val2, SORT_N2, stream);
    gather_obox<<<SORT_N2 / 256, 256, 0, stream>>>(boxes, val2, obox);
    nms_kernel<<<1, 1024, 0, stream>>>(obox, key2, alive);
    gather_proposals<<<1, 256, 0, stream>>>(obox, alive, key2, (float4*)out);

    // labeling + sampling + losses
    int ablocks = (NANCH + 255) / 256;
    iou_pass1<<<ablocks, 256, 0, stream>>>((const float4*)gt_box, m_val, m_idx,
                                           gmax);
    label_pass<<<ablocks, 256, 0, stream>>>((const float4*)gt_box, gmax, m_val,
                                            label, posCount);
    build_sample_keys<<<SORT_N1 / 256, 256, 0, stream>>>(smp_rand, label, keyP,
                                                         valP, keyN, valN);
    bitonic_sort(keyP, valP, SORT_N1, stream);
    bitonic_sort(keyN, valN, SORT_N1, stream);

    loss_kernel<<<1, 384, 0, stream>>>(cls_logits, regression,
                                       (const float4*)gt_box, m_idx, valP,
                                       valN, posCount, out + 8000);
}
